// EncoderNet_13511967113420
// MI455X (gfx1250) — matrix operand order repaired and checked by reading, not compiled, not run on hardware
//
#include <hip/hip_runtime.h>
#include <hip/hip_bf16.h>

typedef __attribute__((ext_vector_type(16))) _Float16 v16h;
typedef __attribute__((ext_vector_type(8)))  _Float16 v8h;
typedef __attribute__((ext_vector_type(8)))  float    v8f;

#define EPSBN 1e-5f

// ---------------------------------------------------------------------------
// Front-end: RelationAwareness (2 maps x 6 heads) + BN2d -> f16 NHWC [64,68,68,12]
// ---------------------------------------------------------------------------
__global__ void front_kernel(const float* __restrict__ x,      // [64,370]
                             const float* __restrict__ loc,    // [62,3]
                             const int*   __restrict__ ro,     // [2,62]
                             const float* __restrict__ dw,     // [5,60]
                             const float* __restrict__ db,     // [60]
                             const float* __restrict__ lw,     // [3,60]
                             const float* __restrict__ lb,     // [60]
                             const float* __restrict__ ew,     // [10,60]
                             const float* __restrict__ eb,     // [60]
                             const float* __restrict__ av,     // [20]
                             const float* __restrict__ bng, const float* __restrict__ bnb,
                             const float* __restrict__ bnm, const float* __restrict__ bnv,
                             _Float16* __restrict__ out)       // [64,68,68,12]
{
    __shared__ float z[68 * 60];
    __shared__ float w1[6 * 68];
    __shared__ float w2[6 * 68];
    const int b   = blockIdx.x;
    const int tid = threadIdx.x;
    const float* xb = x + (size_t)b * 370;

    for (int m = 0; m < 2; ++m) {
        const int* perm = ro + m * 62;
        for (int t = tid; t < 68 * 60; t += 256) {
            int node = t / 60, col = t - node * 60;
            float acc;
            if (node < 62) {
                int src = perm[node];
                acc = db[col] + lb[col];
                #pragma unroll
                for (int f = 0; f < 5; ++f) acc += xb[src * 5 + f] * dw[f * 60 + col];
                #pragma unroll
                for (int d = 0; d < 3; ++d) acc += loc[src * 3 + d] * lw[d * 60 + col];
            } else {
                int k2 = node - 62;
                acc = eb[col];
                #pragma unroll
                for (int f = 0; f < 10; ++f) acc += xb[310 + k2 * 10 + f] * ew[f * 60 + col];
            }
            z[t] = fmaxf(acc, 0.f);
        }
        __syncthreads();
        for (int t = tid; t < 6 * 68; t += 256) {
            int h = t / 68, node = t - h * 68;
            float s1 = 0.f, s2 = 0.f;
            #pragma unroll
            for (int e = 0; e < 10; ++e) {
                float zz = z[node * 60 + h * 10 + e];
                s1 += zz * av[e];
                s2 += zz * av[10 + e];
            }
            w1[t] = s1; w2[t] = s2;
        }
        __syncthreads();
        for (int t = tid; t < 68 * 68 * 6; t += 256) {
            int h = t / (68 * 68);
            int r = t - h * 68 * 68;
            int i = r / 68, j = r - i * 68;
            int c = m * 6 + h;
            float v = w1[h * 68 + i] + w2[h * 68 + j];
            v = bng[c] * (v - bnm[c]) * rsqrtf(bnv[c] + EPSBN) + bnb[c];
            out[(((size_t)b * 68 + i) * 68 + j) * 12 + c] = (_Float16)v;
        }
        __syncthreads();
    }
}

// ---------------------------------------------------------------------------
// Weight prep: OIHW fp32 -> [N, Kp] f16, k = (kh*KW+kw)*Cin + c, zero-padded
// ---------------------------------------------------------------------------
__global__ void prep_conv_w_kernel(const float* __restrict__ w, _Float16* __restrict__ wt,
                                   int N, int Cin, int KH, int KW, int Kp)
{
    int i = blockIdx.x * 256 + threadIdx.x;
    if (i >= N * Kp) return;
    int n = i / Kp, k = i - n * Kp;
    int K = Cin * KH * KW;
    float v = 0.f;
    if (k < K) {
        int p = k / Cin, c = k - p * Cin;
        int kh = p / KW, kw = p - kh * KW;
        v = w[(((size_t)n * Cin + c) * KH + kh) * KW + kw];
    }
    wt[i] = (_Float16)v;
}

// Dense: src [K, N] fp32 -> [N, Kp] f16
__global__ void prep_dense_w_kernel(const float* __restrict__ w, _Float16* __restrict__ wt,
                                    int N, int K, int Kp)
{
    int i = blockIdx.x * 256 + threadIdx.x;
    if (i >= N * Kp) return;
    int n = i / Kp, k = i - n * Kp;
    wt[i] = (_Float16)(k < K ? w[(size_t)k * N + n] : 0.f);
}

// ---------------------------------------------------------------------------
// A-fragment gather for one 16-row M sub-tile (wave32 WMMA layout):
// element e of lane (g,lm): k = kk + g*8 + (e<8 ? e : e+8)
// ---------------------------------------------------------------------------
__device__ __forceinline__ v16h gather_a(
    const _Float16* __restrict__ act, const _Float16* __restrict__ arow,
    int kk, int g, bool onexone, bool chunk3, int clog,
    int ih0, int iw0, int b, int IH, int IW, int Cin, int KW, int Kreal)
{
    v16h af;
    if (onexone) {
        const v8h lo = *(const v8h*)(arow + kk + g * 8);
        const v8h hi = *(const v8h*)(arow + kk + 16 + g * 8);
        #pragma unroll
        for (int e = 0; e < 8; ++e) { af[e] = lo[e]; af[e + 8] = hi[e]; }
    } else if (chunk3) {
        // whole 32-chunk shares one kernel position p = kk/Cin (Cin pow2, mult of 32)
        int p  = kk >> clog;
        int c0 = kk & (Cin - 1);
        int kh = (p * 86) >> 8;        // p/3 exact for 0 <= p <= 8
        int kw = p - 3 * kh;
        int ih = ih0 + kh, iw = iw0 + kw;
        if ((unsigned)ih < (unsigned)IH && (unsigned)iw < (unsigned)IW) {
            const _Float16* src = act + (((size_t)b * IH + ih) * IW + iw) * Cin + c0;
            const v8h lo = *(const v8h*)(src + g * 8);
            const v8h hi = *(const v8h*)(src + 16 + g * 8);
            #pragma unroll
            for (int e = 0; e < 8; ++e) { af[e] = lo[e]; af[e + 8] = hi[e]; }
        } else {
            #pragma unroll
            for (int e = 0; e < 16; ++e) af[e] = (_Float16)0.f;
        }
    } else {
        // general gather (conv1: Cin=12, 7x7)
        #pragma unroll
        for (int e = 0; e < 16; ++e) {
            int k = kk + g * 8 + (e < 8 ? e : e + 8);
            _Float16 v = (_Float16)0.f;
            if (k < Kreal) {
                unsigned p = (unsigned)k / (unsigned)Cin;
                int cix = k - (int)p * Cin;
                unsigned kh = p / (unsigned)KW;
                int kw = (int)p - (int)kh * KW;
                int ih = ih0 + (int)kh, iw = iw0 + kw;
                if ((unsigned)ih < (unsigned)IH && (unsigned)iw < (unsigned)IW)
                    v = act[(((size_t)b * IH + ih) * IW + iw) * Cin + cix];
            }
            af[e] = v;
        }
    }
    return af;
}

__device__ __forceinline__ v16h load_b(const _Float16* __restrict__ wrow, int kk, int g)
{
    // B 32x16 f16 layout: lane half g holds K kk+g*8 .. +7 and kk+16+g*8 .. +7
    const v8h lo = *(const v8h*)(wrow + kk + g * 8);
    const v8h hi = *(const v8h*)(wrow + kk + 16 + g * 8);
    v16h bf;
    #pragma unroll
    for (int e = 0; e < 8; ++e) { bf[e] = lo[e]; bf[e + 8] = hi[e]; }
    return bf;
}

// ---------------------------------------------------------------------------
// Implicit-GEMM conv via v_wmma_f32_16x16x32_f16.
// One wave per 32x32 output tile (2 M sub-tiles x 2 N sub-tiles -> 4 WMMAs
// per 32-wide K-chunk; A and B fragments are each reused twice).
// Epilogue: (+bias) -> BN -> (+res) -> ACT     (BN_AFTER: ACT -> BN)
// ACT: 0 none, 1 relu, 2 leaky(0.1)
// ---------------------------------------------------------------------------
template <int ACT, bool BN, bool RES, bool BN_AFTER>
__global__ __launch_bounds__(32)
void conv_gemm_wmma(const _Float16* __restrict__ act,
                    const _Float16* __restrict__ wt,
                    const float* __restrict__ bng, const float* __restrict__ bnb,
                    const float* __restrict__ bnm, const float* __restrict__ bnv,
                    const float* __restrict__ bias,
                    const _Float16* __restrict__ res,
                    _Float16* __restrict__ out,
                    int IH, int IW, int Cin,
                    int OH, int OW, int KH, int KW, int stride, int pad,
                    int N, int Kreal, int Kp)
{
    const int lane = threadIdx.x & 31;
    const int g    = lane >> 4;        // lane group (0: lanes 0-15, 1: lanes 16-31)
    const int lm   = lane & 15;
    const int m0   = blockIdx.x * 32;
    const int n0   = blockIdx.y * 32;
    const int mrow0 = m0 + lm;         // A row, M sub-tile 0
    const int mrow1 = m0 + 16 + lm;    // A row, M sub-tile 1

    // decompose both rows
    int ow0 = mrow0 % OW; int t0 = mrow0 / OW; int oh0 = t0 % OH; int b0 = t0 / OH;
    int ow1 = mrow1 % OW; int t1 = mrow1 / OW; int oh1 = t1 % OH; int b1 = t1 / OH;
    const int ih0a = oh0 * stride - pad, iw0a = ow0 * stride - pad;
    const int ih0b = oh1 * stride - pad, iw0b = ow1 * stride - pad;

    const bool onexone = (KH == 1) && (KW == 1);               // pad==0 for all 1x1 here
    const bool chunk3  = (!onexone) && (KW == 3) &&
                         (Cin >= 32) && ((Cin & (Cin - 1)) == 0);
    const int  clog    = 31 - __clz(Cin);

    const _Float16* arow0 = act + (((size_t)b0 * IH + ih0a) * IW + iw0a) * Cin; // 1x1 path
    const _Float16* arow1 = act + (((size_t)b1 * IH + ih0b) * IW + iw0b) * Cin;
    const _Float16* wrow0 = wt + (size_t)(n0 + lm) * Kp;
    const _Float16* wrow1 = wt + (size_t)(n0 + 16 + lm) * Kp;

    v8f acc00 = {}, acc01 = {}, acc10 = {}, acc11 = {};
    for (int kk = 0; kk < Kp; kk += 32) {
        const v16h af0 = gather_a(act, arow0, kk, g, onexone, chunk3, clog,
                                  ih0a, iw0a, b0, IH, IW, Cin, KW, Kreal);
        const v16h af1 = gather_a(act, arow1, kk, g, onexone, chunk3, clog,
                                  ih0b, iw0b, b1, IH, IW, Cin, KW, Kreal);
        const v16h bf0 = load_b(wrow0, kk, g);
        const v16h bf1 = load_b(wrow1, kk, g);
        acc00 = __builtin_amdgcn_wmma_f32_16x16x32_f16(false, af0, false, bf0, (short)0, acc00, false, false);
        acc01 = __builtin_amdgcn_wmma_f32_16x16x32_f16(false, af0, false, bf1, (short)0, acc01, false, false);
        acc10 = __builtin_amdgcn_wmma_f32_16x16x32_f16(false, af1, false, bf0, (short)0, acc10, false, false);
        acc11 = __builtin_amdgcn_wmma_f32_16x16x32_f16(false, af1, false, bf1, (short)0, acc11, false, false);
    }

    // Epilogue: C/D layout -> row = m0 + mt*16 + g*8 + r, col = n0 + nt*16 + lm
    #pragma unroll
    for (int nt = 0; nt < 2; ++nt) {
        const int col = n0 + nt * 16 + lm;
        float scale = 1.f, shift = 0.f;
        if (BN) {
            float gg = bng[col], bb = bnb[col], mm = bnm[col], vv = bnv[col];
            scale = gg * rsqrtf(vv + EPSBN);
            shift = bb - mm * scale;
        }
        const float bsv = bias ? bias[col] : 0.f;
        #pragma unroll
        for (int mt = 0; mt < 2; ++mt) {
            const v8f& acc = mt ? (nt ? acc11 : acc10) : (nt ? acc01 : acc00);
            #pragma unroll
            for (int r = 0; r < 8; ++r) {
                int row = m0 + mt * 16 + g * 8 + r;
                float v = acc[r] + bsv;
                if (!BN_AFTER) {
                    if (BN)  v = v * scale + shift;
                    if (RES) v += (float)res[(size_t)row * N + col];
                    if (ACT == 1) v = fmaxf(v, 0.f);
                    else if (ACT == 2) v = v > 0.f ? v : 0.1f * v;
                } else {
                    if (ACT == 1) v = fmaxf(v, 0.f);
                    else if (ACT == 2) v = v > 0.f ? v : 0.1f * v;
                    if (BN) v = v * scale + shift;
                }
                out[(size_t)row * N + col] = (_Float16)v;
            }
        }
    }
}

// ---------------------------------------------------------------------------
// MaxPool 3x3 s2 p1 on [64,34,34,64] -> [64,17,17,64] (f16)
// ---------------------------------------------------------------------------
__global__ void maxpool_kernel(const _Float16* __restrict__ in, _Float16* __restrict__ out)
{
    const int C = 64, IH = 34, IW = 34, OH = 17, OW = 17;
    int i = blockIdx.x * 256 + threadIdx.x;
    if (i >= 64 * OH * OW * C) return;
    int c = i % C; int t = i / C;
    int ow = t % OW; t /= OW;
    int oh = t % OH; int b = t / OH;
    float m = -3.0e38f;
    #pragma unroll
    for (int kh = 0; kh < 3; ++kh)
        #pragma unroll
        for (int kw = 0; kw < 3; ++kw) {
            int ih = oh * 2 - 1 + kh, iw = ow * 2 - 1 + kw;
            if ((unsigned)ih < (unsigned)IH && (unsigned)iw < (unsigned)IW)
                m = fmaxf(m, (float)in[(((size_t)b * IH + ih) * IW + iw) * C + c]);
        }
    out[i] = (_Float16)m;
}

// AvgPool [64,3,3,2048] -> [64,2048] f16
__global__ void avgpool_kernel(const _Float16* __restrict__ in, _Float16* __restrict__ out)
{
    int i = blockIdx.x * 256 + threadIdx.x;
    if (i >= 64 * 2048) return;
    int c = i & 2047, b = i >> 11;
    float s = 0.f;
    #pragma unroll
    for (int y = 0; y < 9; ++y) s += (float)in[((size_t)b * 9 + y) * 2048 + c];
    out[i] = (_Float16)(s * (1.f / 9.f));
}

// Final 512 -> 2 head, fp32 output [64,2]
__global__ void mlp1_kernel(const _Float16* __restrict__ f, const float* __restrict__ w,
                            const float* __restrict__ bias, float* __restrict__ out)
{
    int i = threadIdx.x;
    if (i >= 128) return;
    int b = i >> 1, cl = i & 1;
    float acc = bias[cl];
    for (int k = 0; k < 512; ++k) acc += (float)f[b * 512 + k] * w[k * 2 + cl];
    out[i] = acc;
}

// ---------------------------------------------------------------------------
// Host orchestration
// ---------------------------------------------------------------------------
struct BNp { const float *g = nullptr, *b = nullptr, *m = nullptr, *v = nullptr; };

static inline void launch_gemm(int mode, hipStream_t s,
    const _Float16* act, const _Float16* wt, BNp bn,
    const float* bias, const _Float16* res, _Float16* out,
    int IH, int IW, int Cin, int OH, int OW, int KH, int KW, int stride, int pad,
    int N, int K, int Kp)
{
    int M = 64 * OH * OW;
    dim3 grid(M / 32, N / 32), blk(32, 1, 1);
    switch (mode) {
    case 0: conv_gemm_wmma<1, true,  false, false><<<grid, blk, 0, s>>>(act, wt, bn.g, bn.b, bn.m, bn.v, bias, res, out, IH, IW, Cin, OH, OW, KH, KW, stride, pad, N, K, Kp); break;
    case 1: conv_gemm_wmma<1, true,  true,  false><<<grid, blk, 0, s>>>(act, wt, bn.g, bn.b, bn.m, bn.v, bias, res, out, IH, IW, Cin, OH, OW, KH, KW, stride, pad, N, K, Kp); break;
    case 2: conv_gemm_wmma<0, true,  false, false><<<grid, blk, 0, s>>>(act, wt, bn.g, bn.b, bn.m, bn.v, bias, res, out, IH, IW, Cin, OH, OW, KH, KW, stride, pad, N, K, Kp); break;
    case 3: conv_gemm_wmma<0, false, false, false><<<grid, blk, 0, s>>>(act, wt, bn.g, bn.b, bn.m, bn.v, bias, res, out, IH, IW, Cin, OH, OW, KH, KW, stride, pad, N, K, Kp); break;
    case 4: conv_gemm_wmma<2, true,  false, true ><<<grid, blk, 0, s>>>(act, wt, bn.g, bn.b, bn.m, bn.v, bias, res, out, IH, IW, Cin, OH, OW, KH, KW, stride, pad, N, K, Kp); break;
    }
}

extern "C" void kernel_launch(void* const* d_in, const int* in_sizes, int n_in,
                              void* d_out, int out_size, void* d_ws, size_t ws_size,
                              hipStream_t stream)
{
    (void)in_sizes; (void)n_in; (void)out_size; (void)ws_size;

    // ---- Input walk (jax pytree order: dict keys sorted alphabetically) ----
    int cur = 0;
    auto F = [&](int i) { return (const float*)d_in[i]; };
    auto rdBN = [&]() { BNp r; r.b = F(cur); r.g = F(cur + 1); r.m = F(cur + 2); r.v = F(cur + 3); cur += 4; return r; };

    const float* loc  = F(cur++);       // location
    const float* avec = F(cur++);       // params.a
    BNp bn1d = rdBN();                  // params.bn1d {b,g,m,v}
    BNp bn2d = rdBN();                  // params.bn2d
    const float* data_b = F(cur++); const float* data_w = F(cur++);
    const float* eye_b  = F(cur++); const float* eye_w  = F(cur++);
    const float* loc_b  = F(cur++); const float* loc_w  = F(cur++);
    const float* mlp0_b = F(cur++); const float* mlp0_w = F(cur++);
    const float* mlp1_b = F(cur++); const float* mlp1_w = F(cur++);
    // params.resnet: bn1, conv1, fc_b, fc_w, layers
    BNp rbn1 = rdBN();
    const float* conv1w = F(cur++);
    const float* fc_b   = F(cur++);
    const float* fc_w   = F(cur++);
    BNp Lbn1[16], Lbn2[16], Lbn3[16], Ldbn[16];
    const float *Ldw[16], *Lw1[16], *Lw2[16], *Lw3[16];
    bool Lhas[16];
    const int nb[4] = {3, 4, 6, 3};
    {
        int bi = 0;
        for (int s = 0; s < 4; ++s)
            for (int j = 0; j < nb[s]; ++j, ++bi) {
                Lbn1[bi] = rdBN(); Lbn2[bi] = rdBN(); Lbn3[bi] = rdBN();
                if (j == 0) { Ldbn[bi] = rdBN(); Ldw[bi] = F(cur++); Lhas[bi] = true; }
                else        { Ldw[bi] = nullptr; Lhas[bi] = false; }
                Lw1[bi] = F(cur++); Lw2[bi] = F(cur++); Lw3[bi] = F(cur++);
            }
    }
    const int*   rorder = (const int*)d_in[cur++];  // rand_order
    const float* x      = F(cur++);                 // x

    // ---- Workspace layout: 6 activation regions (5M halfs) + weight scratch ----
    const size_t REG = 5u * 1024u * 1024u;  // halfs per region
    char* base = (char*)d_ws;
    _Float16* R[6];
    for (int i = 0; i < 6; ++i) R[i] = (_Float16*)(base + (size_t)i * REG * 2);
    _Float16* WREG = (_Float16*)(base + (size_t)6 * REG * 2);  // 2.5M halfs used max

    auto prep_conv = [&](const float* w, int N, int Cin, int KH, int KW, int Kp) {
        int tot = N * Kp;
        prep_conv_w_kernel<<<(tot + 255) / 256, 256, 0, stream>>>(w, WREG, N, Cin, KH, KW, Kp);
    };
    auto prep_dense = [&](const float* w, int N, int K, int Kp) {
        int tot = N * Kp;
        prep_dense_w_kernel<<<(tot + 255) / 256, 256, 0, stream>>>(w, WREG, N, K, Kp);
    };

    // ---- Front-end -> conv1 input f16 NHWC [64,68,68,12] in R0 ----
    front_kernel<<<64, 256, 0, stream>>>(x, loc, rorder, data_w, data_b, loc_w, loc_b,
                                         eye_w, eye_b, avec,
                                         bn2d.g, bn2d.b, bn2d.m, bn2d.v, R[0]);

    // ---- conv1: 7x7 s2 p3, 12->64, 68->34 (K=588, Kp=608) ----
    prep_conv(conv1w, 64, 12, 7, 7, 608);
    launch_gemm(0, stream, R[0], WREG, rbn1, nullptr, nullptr, R[1],
                68, 68, 12, 34, 34, 7, 7, 2, 3, 64, 588, 608);

    // ---- maxpool 3x3 s2 p1 -> [64,17,17,64] in R2 ----
    maxpool_kernel<<<(64 * 17 * 17 * 64 + 255) / 256, 256, 0, stream>>>(R[1], R[2]);

    // ---- ResNet-50 stages ----
    _Float16* curp = R[2];
    int inC = 64, H = 17, outsel = 0;
    const int planes[4] = {64, 128, 256, 512};
    int bi = 0;
    for (int s = 0; s < 4; ++s) {
        int P = planes[s];
        for (int j = 0; j < nb[s]; ++j, ++bi) {
            int st = (j == 0 && s > 0) ? 2 : 1;
            int Ho = (st == 2) ? (H + 2 - 3) / 2 + 1 : H;
            // conv1 1x1 s1 + BN + relu -> R3
            prep_conv(Lw1[bi], P, inC, 1, 1, inC);
            launch_gemm(0, stream, curp, WREG, Lbn1[bi], nullptr, nullptr, R[3],
                        H, H, inC, H, H, 1, 1, 1, 0, P, inC, inC);
            // conv2 3x3 stride st pad1 + BN + relu -> R4
            prep_conv(Lw2[bi], P, P, 3, 3, 9 * P);
            launch_gemm(0, stream, R[3], WREG, Lbn2[bi], nullptr, nullptr, R[4],
                        H, H, P, Ho, Ho, 3, 3, st, 1, P, 9 * P, 9 * P);
            // identity path
            const _Float16* resp = curp;
            if (Lhas[bi]) {
                prep_conv(Ldw[bi], 4 * P, inC, 1, 1, inC);
                launch_gemm(2, stream, curp, WREG, Ldbn[bi], nullptr, nullptr, R[5],
                            H, H, inC, Ho, Ho, 1, 1, st, 0, 4 * P, inC, inC);
                resp = R[5];
            }
            // conv3 1x1 + BN + residual + relu -> out
            _Float16* outp = outsel ? R[1] : R[0];
            outsel ^= 1;
            prep_conv(Lw3[bi], 4 * P, P, 1, 1, P);
            launch_gemm(1, stream, R[4], WREG, Lbn3[bi], nullptr, resp, outp,
                        Ho, Ho, P, Ho, Ho, 1, 1, 1, 0, 4 * P, P, P);
            curp = outp; inC = 4 * P; H = Ho;
        }
    }

    // ---- head ----
    avgpool_kernel<<<(64 * 2048 + 255) / 256, 256, 0, stream>>>(curp, R[3]);

    // fc: [64,2048] @ [2048,512] + fc_b -> R4
    prep_dense(fc_w, 512, 2048, 2048);
    launch_gemm(3, stream, R[3], WREG, BNp{}, fc_b, nullptr, R[4],
                1, 1, 2048, 1, 1, 1, 1, 1, 0, 512, 2048, 2048);

    // mlp0: leaky(0.1) then bn1d -> R5
    prep_dense(mlp0_w, 512, 512, 512);
    launch_gemm(4, stream, R[4], WREG, bn1d, mlp0_b, nullptr, R[5],
                1, 1, 512, 1, 1, 1, 1, 1, 0, 512, 512, 512);

    // mlp1: 512 -> 2, fp32 out
    mlp1_kernel<<<1, 128, 0, stream>>>(R[5], mlp1_w, mlp1_b, (float*)d_out);
}
